// DSTP_rnn_27573690040441
// MI455X (gfx1250) — compile-verified, matrix operand order checked by reading, not run on hardware
//
#include <hip/hip_runtime.h>
#include <hip/hip_bf16.h>

typedef __attribute__((ext_vector_type(16))) _Float16 v16h;
typedef __attribute__((ext_vector_type(8)))  _Float16 v8h;
typedef __attribute__((ext_vector_type(8)))  float    v8f;

#define TB    16
#define BATCH 2048
#define TM1   9
#define NH    128

// packed weight tile regions (units: _Float16 elements). Each tile = 32 lanes * 16 halfs.
#define PK_W2    0            // encoder LSTM2 [Wih2;Whh2]: K=256,N=512 -> 8*32 = 256 tiles
#define PK_WE    (256*512)    // W_e (d_attn_w1 cols 256:384): K=128,N=128 -> 4*8 = 32 tiles
#define PK_WDC   (288*512)    // [W_d;W_c] (d_attn_w1 cols 0:256): K=256,N=128 -> 8*8 = 64 tiles
#define PK_WDEC  (352*512)    // d_Whh: K=128,N=512 -> 4*32 = 128 tiles
#define PK_TOTAL (480*512)

// -------- LDS layout (bytes) --------
#define OFF_ACT    0        // f16 [16][256]   8192
#define OFF_GATE   8192     // f32 [16][512]   32768
#define OFF_H      40960    // f32 [16][128]   8192
#define OFF_C      49152    // f32 [16][128]   8192
#define OFF_ALPHA  57344    // f32 [16][128]   8192
#define OFF_XENC   65536    // f16 [16][9][128] 36864
#define OFF_E      102400   // f32 [16][9][128] 73728
#define OFF_U      176128   // f32 [16][128]   8192
#define OFF_CTX    184320   // f32 [16][128]   8192
#define OFF_YT     192512   // f32 [16]        64
#define OFF_BIAS2  192576   // f32 [512]
#define OFF_B1     194624   // f32 [128]
#define OFF_W2S    195136   // f32 [128]
#define OFF_DWIH   195648   // f32 [512]
#define OFF_DBIAS  197696   // f32 [512]
#define OFF_FCW    199744   // f32 [129] (pad)
#define SMEM_BYTES 200320

__device__ __forceinline__ float sigm(float x) { return 1.f / (1.f + __expf(-x)); }

__device__ __forceinline__ float wave_sum(float v) {
  #pragma unroll
  for (int off = 16; off; off >>= 1) v += __shfl_xor(v, off, 32);
  return v;
}
__device__ __forceinline__ float wave_max(float v) {
  #pragma unroll
  for (int off = 16; off; off >>= 1) v = fmaxf(v, __shfl_xor(v, off, 32));
  return v;
}

// A fragment loader: per-lane two contiguous 8xf16 chunks (k offsets +0 and +16)
__device__ __forceinline__ v16h load_a(const _Float16* p) {
  v8h lo = *(const v8h*)(p);
  v8h hi = *(const v8h*)(p + 16);
  v16h a;
  #pragma unroll
  for (int q = 0; q < 8; ++q) { a[q] = lo[q]; a[q + 8] = hi[q]; }
  return a;
}

__device__ __forceinline__ v8f wmma_step(v16h a, v16h b, v8f c) {
  return __builtin_amdgcn_wmma_f32_16x16x32_f16(false, a, false, b, (short)0, c, false, false);
}

// ---------------- weight pack kernel: fp32 -> f16 WMMA B-fragment layout ----------------
__global__ void pack_weights(const float* __restrict__ eWih2, const float* __restrict__ eWhh2,
                             const float* __restrict__ dW1,   const float* __restrict__ dWhh,
                             _Float16* __restrict__ packed) {
  int gid = blockIdx.x * blockDim.x + threadIdx.x;
  if (gid >= PK_TOTAL) return;
  int tile   = gid >> 9;
  int within = gid & 511;
  int lane = within >> 4;
  int j    = within & 15;
  int v = j >> 1, hh = j & 1, g = lane >> 4, nl = lane & 15;
  int kl = (v < 4 ? 0 : 16) + g * 8 + (v & 3) * 2 + hh;   // B[k][n], n = nl
  float val;
  if (tile < 256) {                       // W2cat: B[k][n] = k<128 ? Wih2[n][k] : Whh2[n][k-128]
    int kt = tile >> 5, nt = tile & 31;
    int k = kt * 32 + kl, n = nt * 16 + nl;
    val = (k < 128) ? eWih2[n * 128 + k] : eWhh2[n * 128 + (k - 128)];
  } else if (tile < 288) {                // We: B[k][n] = d_attn_w1[n][256+k]
    int ti = tile - 256; int kt = ti >> 3, nt = ti & 7;
    int k = kt * 32 + kl, n = nt * 16 + nl;
    val = dW1[n * 384 + 256 + k];
  } else if (tile < 352) {                // Wdc: B[k][n] = d_attn_w1[n][k]
    int ti = tile - 288; int kt = ti >> 3, nt = ti & 7;
    int k = kt * 32 + kl, n = nt * 16 + nl;
    val = dW1[n * 384 + k];
  } else {                                // Wdec: B[k][n] = d_Whh[n][k]
    int ti = tile - 352; int kt = ti >> 5, nt = ti & 31;
    int k = kt * 32 + kl, n = nt * 16 + nl;
    val = dWhh[n * 128 + k];
  }
  packed[gid] = (_Float16)val;
}

// ---------------- fused DSTP-RNN kernel: one block = 16 batch rows, all 9+9 steps ----------------
__global__ __launch_bounds__(256, 1)
void dstp_main(const float* __restrict__ X, const float* __restrict__ y_prev,
               const float* __restrict__ e_attn_w,
               const float* __restrict__ e_bih2, const float* __restrict__ e_bhh2,
               const float* __restrict__ d_attn_b1, const float* __restrict__ d_attn_w2,
               const float* __restrict__ d_attn_b2,
               const float* __restrict__ d_Wih, const float* __restrict__ d_bih,
               const float* __restrict__ d_bhh,
               const float* __restrict__ fc_w, const float* __restrict__ fc_b,
               const float* __restrict__ fp_w, const float* __restrict__ fp_b,
               const float* __restrict__ ft_w, const float* __restrict__ ft_b,
               const float* __restrict__ ftr_w, const float* __restrict__ ftr_b,
               const _Float16* __restrict__ packed, float* __restrict__ out) {
  extern __shared__ char smem[];
  _Float16* sAct   = (_Float16*)(smem + OFF_ACT);
  float*    sGate  = (float*)(smem + OFF_GATE);
  float*    sH     = (float*)(smem + OFF_H);
  float*    sC     = (float*)(smem + OFF_C);
  float*    sAlpha = (float*)(smem + OFF_ALPHA);
  _Float16* sXenc  = (_Float16*)(smem + OFF_XENC);
  float*    sE     = (float*)(smem + OFF_E);
  float*    sU     = (float*)(smem + OFF_U);
  float*    sCtx   = (float*)(smem + OFF_CTX);
  float*    sYt    = (float*)(smem + OFF_YT);
  float*    sBias2 = (float*)(smem + OFF_BIAS2);
  float*    sB1    = (float*)(smem + OFF_B1);
  float*    sW2    = (float*)(smem + OFF_W2S);
  float*    sDwih  = (float*)(smem + OFF_DWIH);
  float*    sDbias = (float*)(smem + OFF_DBIAS);
  float*    sFcw   = (float*)(smem + OFF_FCW);

  const int tid  = threadIdx.x;
  const int lane = tid & 31;
  const int wave = tid >> 5;
  const int b0   = blockIdx.x * TB;
  const int mB   = (lane >> 4) * 8;   // C/D fragment row base
  const int nC   = lane & 15;         // C/D fragment col

  // ---- stage small vectors; zero LSTM2 state ----
  for (int e = tid; e < 512; e += 256) {
    sBias2[e] = e_bih2[e] + e_bhh2[e];
    sDbias[e] = d_bih[e] + d_bhh[e];
    sDwih[e]  = d_Wih[e];
  }
  if (tid < 128) { sB1[tid] = d_attn_b1[tid]; sW2[tid] = d_attn_w2[tid]; }
  if (tid < 129) sFcw[tid] = fc_w[tid];
  for (int e = tid; e < TB * NH; e += 256) { sH[e] = 0.f; sC[e] = 0.f; }
  __syncthreads();

  // ---- input attention: alpha(b,i) = softmax_i( sum_t wx[t]*X[b,t,i] ); time-invariant, exact ----
  for (int e = tid; e < TB * NH; e += 256) {
    int row = e >> 7, i = e & 127;
    const float* xp = X + (size_t)(b0 + row) * (TM1 * NH) + i;
    float p = 0.f;
    #pragma unroll
    for (int t = 0; t < TM1; ++t) p += e_attn_w[256 + t] * xp[t * NH];
    sAlpha[e] = p;
  }
  __syncthreads();
  #pragma unroll
  for (int rr = 0; rr < 2; ++rr) {
    int row = wave * 2 + rr;
    float v[4];
    #pragma unroll
    for (int q = 0; q < 4; ++q) v[q] = sAlpha[row * NH + lane + q * 32];
    float m = wave_max(fmaxf(fmaxf(v[0], v[1]), fmaxf(v[2], v[3])));
    float s = 0.f;
    #pragma unroll
    for (int q = 0; q < 4; ++q) { v[q] = __expf(v[q] - m); s += v[q]; }
    float inv = 1.f / wave_sum(s);
    #pragma unroll
    for (int q = 0; q < 4; ++q) sAlpha[row * NH + lane + q * 32] = v[q] * inv;
  }
  __syncthreads();

  // ---- encoder LSTM2, 9 steps (input x_tilde2 = alpha*x_t/128; stage-2 softmax uniform, exact) ----
  #pragma unroll 1
  for (int t = 0; t < TM1; ++t) {
    for (int e = tid; e < TB * 256; e += 256) {
      int row = e >> 8, f = e & 255;
      float val = (f < 128)
          ? sAlpha[row * NH + f] * X[(size_t)(b0 + row) * (TM1 * NH) + t * NH + f] * (1.f / 128.f)
          : sH[row * NH + (f - 128)];
      sAct[e] = (_Float16)val;
    }
    __syncthreads();
    // gates = [x2|h] @ [Wih2;Whh2]^T : [16x256]x[256x512]
    #pragma unroll
    for (int wi = 0; wi < 4; ++wi) {
      int nt = wave + wi * 8;
      v8f acc = {};
      #pragma unroll
      for (int kt = 0; kt < 8; ++kt) {
        v16h a = load_a(sAct + (lane & 15) * 256 + kt * 32 + (lane >> 4) * 8);
        v16h b = *(const v16h*)(packed + PK_W2 + (size_t)(kt * 32 + nt) * 512 + lane * 16);
        acc = wmma_step(a, b, acc);
      }
      #pragma unroll
      for (int r = 0; r < 8; ++r) sGate[(mB + r) * 512 + nt * 16 + nC] = acc[r];
    }
    __syncthreads();
    for (int e = tid; e < TB * NH; e += 256) {
      int row = e >> 7, j = e & 127;
      float gi = sGate[row * 512 + j]       + sBias2[j];
      float gf = sGate[row * 512 + 128 + j] + sBias2[128 + j];
      float gg = sGate[row * 512 + 256 + j] + sBias2[256 + j];
      float go = sGate[row * 512 + 384 + j] + sBias2[384 + j];
      float c = sigm(gf) * sC[e] + sigm(gi) * tanhf(gg);
      float h = sigm(go) * tanhf(c);
      sC[e] = c; sH[e] = h;
      sXenc[(row * TM1 + t) * NH + j] = (_Float16)h;
    }
    __syncthreads();
  }

  // ---- E(b,tau,:) = X_enc(tau) @ We^T + b1 (step-invariant part of decoder attention) ----
  #pragma unroll 1
  for (int it = 0; it < 9; ++it) {
    int tt = wave * 9 + it;           // 72 tiles: tau(9) x nt(8)
    int tau = tt >> 3, nt = tt & 7;
    v8f acc = {};
    #pragma unroll
    for (int kt = 0; kt < 4; ++kt) {
      v16h a = load_a(sXenc + ((lane & 15) * TM1 + tau) * NH + kt * 32 + (lane >> 4) * 8);
      v16h b = *(const v16h*)(packed + PK_WE + (size_t)(kt * 8 + nt) * 512 + lane * 16);
      acc = wmma_step(a, b, acc);
    }
    #pragma unroll
    for (int r = 0; r < 8; ++r)
      sE[((mB + r) * TM1 + tau) * NH + nt * 16 + nC] = acc[r] + sB1[nt * 16 + nC];
  }
  for (int e = tid; e < TB * NH; e += 256) { sH[e] = 0.f; sC[e] = 0.f; }  // decoder state
  __syncthreads();

  // ---- decoder, 9 steps ----
  const float b2  = d_attn_b2[0];
  const float fcb = fc_b[0];
  #pragma unroll 1
  for (int t = 0; t < TM1; ++t) {
    for (int e = tid; e < TB * 256; e += 256) {
      int row = e >> 8, f = e & 255;
      float val = (f < 128) ? sH[row * NH + f] : sC[row * NH + (f - 128)];
      sAct[e] = (_Float16)val;
    }
    __syncthreads();
    // u = [d|c] @ [W_d;W_c]^T : [16x256]x[256x128]
    {
      int nt = wave;
      v8f acc = {};
      #pragma unroll
      for (int kt = 0; kt < 8; ++kt) {
        v16h a = load_a(sAct + (lane & 15) * 256 + kt * 32 + (lane >> 4) * 8);
        v16h b = *(const v16h*)(packed + PK_WDC + (size_t)(kt * 8 + nt) * 512 + lane * 16);
        acc = wmma_step(a, b, acc);
      }
      #pragma unroll
      for (int r = 0; r < 8; ++r) sU[(mB + r) * NH + nt * 16 + nC] = acc[r];
    }
    __syncthreads();
    // temporal attention (softmax over 9), context, y_tilde — wave handles 2 rows
    #pragma unroll
    for (int rr = 0; rr < 2; ++rr) {
      int row = wave * 2 + rr;
      float sc[TM1];
      #pragma unroll
      for (int tau = 0; tau < TM1; ++tau) {
        float p = 0.f;
        #pragma unroll
        for (int q = 0; q < 4; ++q) {
          int n = lane + q * 32;
          p += tanhf(sE[(row * TM1 + tau) * NH + n] + sU[row * NH + n]) * sW2[n];
        }
        sc[tau] = wave_sum(p) + b2;
      }
      float m = sc[0];
      #pragma unroll
      for (int tau = 1; tau < TM1; ++tau) m = fmaxf(m, sc[tau]);
      float ssum = 0.f;
      #pragma unroll
      for (int tau = 0; tau < TM1; ++tau) { sc[tau] = __expf(sc[tau] - m); ssum += sc[tau]; }
      float inv = 1.f / ssum;
      float p = 0.f;
      #pragma unroll
      for (int q = 0; q < 4; ++q) {
        int n = lane + q * 32;
        float acc = 0.f;
        #pragma unroll
        for (int tau = 0; tau < TM1; ++tau)
          acc += sc[tau] * inv * (float)sXenc[(row * TM1 + tau) * NH + n];
        sCtx[row * NH + n] = acc;
        p += sFcw[n] * acc;
      }
      p = wave_sum(p);
      if (lane == 0)
        sYt[row] = p + sFcw[128] * y_prev[(size_t)(b0 + row) * TM1 + t] + fcb;
    }
    __syncthreads();
    // gates_h = d @ d_Whh^T : [16x128]x[128x512]  (A = first 128 features of sAct)
    #pragma unroll
    for (int wi = 0; wi < 4; ++wi) {
      int nt = wave + wi * 8;
      v8f acc = {};
      #pragma unroll
      for (int kt = 0; kt < 4; ++kt) {
        v16h a = load_a(sAct + (lane & 15) * 256 + kt * 32 + (lane >> 4) * 8);
        v16h b = *(const v16h*)(packed + PK_WDEC + (size_t)(kt * 32 + nt) * 512 + lane * 16);
        acc = wmma_step(a, b, acc);
      }
      #pragma unroll
      for (int r = 0; r < 8; ++r) sGate[(mB + r) * 512 + nt * 16 + nC] = acc[r];
    }
    __syncthreads();
    for (int e = tid; e < TB * NH; e += 256) {
      int row = e >> 7, j = e & 127;
      float yt = sYt[row];
      float gi = sGate[row * 512 + j]       + sDbias[j]       + yt * sDwih[j];
      float gf = sGate[row * 512 + 128 + j] + sDbias[128 + j] + yt * sDwih[128 + j];
      float gg = sGate[row * 512 + 256 + j] + sDbias[256 + j] + yt * sDwih[256 + j];
      float go = sGate[row * 512 + 384 + j] + sDbias[384 + j] + yt * sDwih[384 + j];
      float c = sigm(gf) * sC[e] + sigm(gi) * tanhf(gg);
      float h = sigm(go) * tanhf(c);
      sC[e] = c; sH[e] = h;
    }
    __syncthreads();
  }

  // ---- heads: final = [d_fin | context_last]; price + two 3-way softmaxes ----
  #pragma unroll
  for (int rr = 0; rr < 2; ++rr) {
    int row = wave * 2 + rr;
    int b = b0 + row;
    float fv[8];
    #pragma unroll
    for (int q = 0; q < 8; ++q) {
      int j = lane * 8 + q;
      fv[q] = (j < 128) ? sH[row * NH + j] : sCtx[row * NH + (j - 128)];
    }
    auto dotw = [&](const float* w) {
      float p = 0.f;
      #pragma unroll
      for (int q = 0; q < 8; ++q) p += w[lane * 8 + q] * fv[q];
      return wave_sum(p);
    };
    float price = dotw(fp_w) + fp_b[0];
    float t0 = dotw(ft_w)        + ft_b[0];
    float t1 = dotw(ft_w + 256)  + ft_b[1];
    float t2 = dotw(ft_w + 512)  + ft_b[2];
    float r0 = dotw(ftr_w)       + ftr_b[0];
    float r1 = dotw(ftr_w + 256) + ftr_b[1];
    float r2 = dotw(ftr_w + 512) + ftr_b[2];
    if (lane == 0) {
      out[b] = price;
      float tm = fmaxf(t0, fmaxf(t1, t2));
      float e0 = __expf(t0 - tm), e1 = __expf(t1 - tm), e2 = __expf(t2 - tm);
      float ti = 1.f / (e0 + e1 + e2);
      out[2048 + b * 3 + 0] = e0 * ti;
      out[2048 + b * 3 + 1] = e1 * ti;
      out[2048 + b * 3 + 2] = e2 * ti;
      float rm = fmaxf(r0, fmaxf(r1, r2));
      float f0 = __expf(r0 - rm), f1 = __expf(r1 - rm), f2 = __expf(r2 - rm);
      float ri = 1.f / (f0 + f1 + f2);
      out[8192 + b * 3 + 0] = f0 * ri;
      out[8192 + b * 3 + 1] = f1 * ri;
      out[8192 + b * 3 + 2] = f2 * ri;
    }
  }
}

extern "C" void kernel_launch(void* const* d_in, const int* in_sizes, int n_in,
                              void* d_out, int out_size, void* d_ws, size_t ws_size,
                              hipStream_t stream) {
  const float* X         = (const float*)d_in[0];
  const float* y_prev    = (const float*)d_in[1];
  const float* e_attn_w  = (const float*)d_in[2];
  const float* e_Wih2    = (const float*)d_in[10];
  const float* e_Whh2    = (const float*)d_in[11];
  const float* e_bih2    = (const float*)d_in[12];
  const float* e_bhh2    = (const float*)d_in[13];
  const float* d_attn_w1 = (const float*)d_in[14];
  const float* d_attn_b1 = (const float*)d_in[15];
  const float* d_attn_w2 = (const float*)d_in[16];
  const float* d_attn_b2 = (const float*)d_in[17];
  const float* d_Wih_p   = (const float*)d_in[18];
  const float* d_Whh_p   = (const float*)d_in[19];
  const float* d_bih_p   = (const float*)d_in[20];
  const float* d_bhh_p   = (const float*)d_in[21];
  const float* fc_w      = (const float*)d_in[22];
  const float* fc_b      = (const float*)d_in[23];
  const float* fp_w      = (const float*)d_in[24];
  const float* fp_b      = (const float*)d_in[25];
  const float* ft_w      = (const float*)d_in[26];
  const float* ft_b      = (const float*)d_in[27];
  const float* ftr_w     = (const float*)d_in[28];
  const float* ftr_b     = (const float*)d_in[29];
  _Float16* packed = (_Float16*)d_ws;
  float* out = (float*)d_out;

  (void)hipFuncSetAttribute(reinterpret_cast<const void*>(dstp_main),
                            hipFuncAttributeMaxDynamicSharedMemorySize, SMEM_BYTES);

  pack_weights<<<(PK_TOTAL + 255) / 256, 256, 0, stream>>>(e_Wih2, e_Whh2, d_attn_w1, d_Whh_p, packed);
  dstp_main<<<BATCH / TB, 256, SMEM_BYTES, stream>>>(
      X, y_prev, e_attn_w, e_bih2, e_bhh2,
      d_attn_b1, d_attn_w2, d_attn_b2, d_Wih_p, d_bih_p, d_bhh_p,
      fc_w, fc_b, fp_w, fp_b, ft_w, ft_b, ftr_w, ftr_b, packed, out);
}
